// Output_66194035966310
// MI455X (gfx1250) — compile-verified, matrix operand order checked by reading, not run on hardware
//
#include <hip/hip_runtime.h>

typedef __attribute__((ext_vector_type(16))) _Float16 v16h;
typedef __attribute__((ext_vector_type(8)))  _Float16 v8h;
typedef __attribute__((ext_vector_type(8)))  float    v8f;

#define SYN_TH 0.8f

// y = x @ binarize(W)^T ; y -= rowmean(y)
// x: [B,256] f32, W: [256,256] f32, out: [B,256] f32
// Block = 256 threads (8 wave32). Wave computes 16 rows x 256 cols.
// K=256 -> 8 chunks of 32 -> v_wmma_f32_16x16x32_f16, f32 accumulate.
// Binarized f16 weights live in LDS in pre-swizzled B-fragment layout:
// fragment (kc, nt) = 32 lanes * 16 halves, lane-contiguous 32B each
// -> two conflict-free ds_load_b128 per fragment per lane.
__global__ __launch_bounds__(256, 1)
void synth_bin_linear(const float* __restrict__ x,
                      const float* __restrict__ w,
                      float* __restrict__ out)
{
    extern __shared__ _Float16 lds[];   // 8 kc * 16 nt * 32 lanes * 16 halves = 128 KB

    const int tid  = threadIdx.x;
    const int lane = tid & 31;
    const int wave = tid >> 5;

    // ---- Phase 1: binarize W into LDS in WMMA B-fragment layout ----
    // B[k][n] = Wbin[n][k]; lane (n&15)+((k&8)?16:0) of frag (k>>5, n>>4)
    // holds W row n, k-runs [k..k+8) at half-offset ((k&16)?8:0).
    // Group g = i*256 + tid ; row n = g>>5 ; k = (g&31)*8  -> a wave reads
    // one full 1 KB row of W per iteration (perfectly coalesced).
    #pragma unroll
    for (int i = 0; i < 32; ++i) {
        const int n = i * 8 + wave;          // 0..255
        const int k = (tid & 31) * 8;        // 0..248
        const float4 f0 = *(const float4*)(w + n * 256 + k);
        const float4 f1 = *(const float4*)(w + n * 256 + k + 4);
        v8h h;
        h[0] = (f0.x > SYN_TH) ? (_Float16)1.0f : (_Float16)0.0f;
        h[1] = (f0.y > SYN_TH) ? (_Float16)1.0f : (_Float16)0.0f;
        h[2] = (f0.z > SYN_TH) ? (_Float16)1.0f : (_Float16)0.0f;
        h[3] = (f0.w > SYN_TH) ? (_Float16)1.0f : (_Float16)0.0f;
        h[4] = (f1.x > SYN_TH) ? (_Float16)1.0f : (_Float16)0.0f;
        h[5] = (f1.y > SYN_TH) ? (_Float16)1.0f : (_Float16)0.0f;
        h[6] = (f1.z > SYN_TH) ? (_Float16)1.0f : (_Float16)0.0f;
        h[7] = (f1.w > SYN_TH) ? (_Float16)1.0f : (_Float16)0.0f;
        const int kc = k >> 5;
        const int kk = k & 31;
        const int fl = (n & 15) | ((kk & 8) ? 16 : 0);
        const int nt = n >> 4;
        const int off = (((kc * 16 + nt) * 32 + fl) << 4) + ((kk & 16) ? 8 : 0);
        *(v8h*)(lds + off) = h;              // 16B-aligned ds_store_b128
    }
    __syncthreads();

    // ---- Phase 2: WMMA GEMM, 16 rows x 256 cols per wave ----
    const int rowBase = blockIdx.x * 128 + wave * 16;
    const int hi      = (lane & 16) ? 1 : 0;
    const int kLane   = hi ? 8 : 0;          // A layout: low lanes K 0-7/16-23, high K 8-15/24-31
    const float* xrow = x + (size_t)(rowBase + (lane & 15)) * 256;

    v8f acc[16];
    const v8f vzero = {0.f, 0.f, 0.f, 0.f, 0.f, 0.f, 0.f, 0.f};
    #pragma unroll
    for (int nt = 0; nt < 16; ++nt) acc[nt] = vzero;

    for (int kc = 0; kc < 8; ++kc) {
        const int k0 = kc * 32 + kLane;
        const float4 a0 = *(const float4*)(xrow + k0);
        const float4 a1 = *(const float4*)(xrow + k0 + 4);
        const float4 a2 = *(const float4*)(xrow + k0 + 16);
        const float4 a3 = *(const float4*)(xrow + k0 + 20);
        v16h A;
        A[0]  = (_Float16)a0.x; A[1]  = (_Float16)a0.y;
        A[2]  = (_Float16)a0.z; A[3]  = (_Float16)a0.w;
        A[4]  = (_Float16)a1.x; A[5]  = (_Float16)a1.y;
        A[6]  = (_Float16)a1.z; A[7]  = (_Float16)a1.w;
        A[8]  = (_Float16)a2.x; A[9]  = (_Float16)a2.y;
        A[10] = (_Float16)a2.z; A[11] = (_Float16)a2.w;
        A[12] = (_Float16)a3.x; A[13] = (_Float16)a3.y;
        A[14] = (_Float16)a3.z; A[15] = (_Float16)a3.w;

        const v16h* bbase = (const v16h*)lds + (kc * 16) * 32 + lane;
        #pragma unroll
        for (int nt = 0; nt < 16; ++nt) {
            const v16h Bf = bbase[nt * 32];  // 2x ds_load_b128, conflict-free
            acc[nt] = __builtin_amdgcn_wmma_f32_16x16x32_f16(
                false, A, false, Bf, (short)0, acc[nt], false, false);
        }
    }

    // ---- Phase 3: per-row mean (16-lane half-wave reduction), subtract, store ----
    // C/D layout: VGPR r, lanes 0-15 -> (row base+r, col nt*16+lane),
    //             lanes 16-31 -> (row base+8+r, col nt*16+lane-16).
    float mean[8];
    #pragma unroll
    for (int r = 0; r < 8; ++r) {
        float v = 0.f;
        #pragma unroll
        for (int nt = 0; nt < 16; ++nt) v += acc[nt][r];
        v += __shfl_xor(v, 1, 32);
        v += __shfl_xor(v, 2, 32);
        v += __shfl_xor(v, 4, 32);
        v += __shfl_xor(v, 8, 32);           // reduces within each 16-lane half
        mean[r] = v * (1.0f / 256.0f);
    }

    float* orow = out + (size_t)(rowBase + hi * 8) * 256 + (lane & 15);
    #pragma unroll
    for (int nt = 0; nt < 16; ++nt) {
        #pragma unroll
        for (int r = 0; r < 8; ++r)
            orow[(size_t)r * 256 + nt * 16] = acc[nt][r] - mean[r];
    }
}

extern "C" void kernel_launch(void* const* d_in, const int* in_sizes, int n_in,
                              void* d_out, int out_size, void* d_ws, size_t ws_size,
                              hipStream_t stream) {
    (void)n_in; (void)out_size; (void)d_ws; (void)ws_size;
    const float* x = (const float*)d_in[0];
    const float* w = (const float*)d_in[1];
    float* out     = (float*)d_out;

    const int B    = in_sizes[0] / 256;      // 131072
    const int grid = B / 128;                // 128 rows per block -> 1024 blocks
    const size_t shmem = (size_t)8 * 16 * 32 * 16 * sizeof(_Float16); // 131072 B

    hipFuncSetAttribute((const void*)synth_bin_linear,
                        hipFuncAttributeMaxDynamicSharedMemorySize, (int)shmem);
    synth_bin_linear<<<grid, 256, shmem, stream>>>(x, w, out);
}
